// QANetAttBlock_66984309949170
// MI455X (gfx1250) — compile-verified, matrix operand order checked by reading, not run on hardware
//
#include <hip/hip_runtime.h>

#define B_ 4
#define S_ 2048
#define H_ 512
#define NH_ 8
#define DK_ 64

constexpr float LN_EPS_ = 1e-5f;
constexpr float NEG_INF_ = -1e30f;

typedef __attribute__((ext_vector_type(16))) __bf16 v16bf;
typedef __attribute__((ext_vector_type(8)))  __bf16 v8bf;
typedef __attribute__((ext_vector_type(8)))  float  v8f;
typedef __attribute__((ext_vector_type(4)))  float  v4f;

__device__ __forceinline__ v8f zero_v8f() {
  v8f z;
#pragma unroll
  for (int i = 0; i < 8; ++i) z[i] = 0.f;
  return z;
}

// A-style fragment: 16x32 bf16 tile from a row-major matrix (ld elements/row).
// lane (l&15) = row; element K pattern: e<8 -> 8g+e, e>=8 -> 8g+16+(e-8).
// Two aligned 16-byte loads per lane.
__device__ __forceinline__ v16bf load_frag_rm(const __bf16* m, long row0, int ld, int col0) {
  int l = threadIdx.x & 31;
  const __bf16* p = m + (row0 + (l & 15)) * (long)ld + col0 + 8 * (l >> 4);
  v8bf lo = *(const v8bf*)p;
  v8bf hi = *(const v8bf*)(p + 16);
  return __builtin_shufflevector(lo, hi, 0, 1, 2, 3, 4, 5, 6, 7,
                                 8, 9, 10, 11, 12, 13, 14, 15);
}

// Packed fragment: one 32-byte contiguous load per lane.
__device__ __forceinline__ v16bf load_frag_pk(const __bf16* pk, int f) {
  int l = threadIdx.x & 31;
  return *(const v16bf*)(pk + ((long)f * 32 + l) * 16);
}

__device__ __forceinline__ v8f wmma_bf16(v16bf a, v16bf b, v8f c) {
  return __builtin_amdgcn_wmma_f32_16x16x32_bf16(false, a, false, b, (short)0, c,
                                                 false, false);
}

// ---------------- kernel 0: pack fp32 weights into bf16 B-fragments ----------------
// Input: nmat row-major KxN matrices (nmat from grid). Output: per fragment
// (kc in K/32, nt in N/16): pk[(fid*32 + lane)*16 + e] with the WMMA B layout.
// grid.x = nmat * (K/32) * (N/16), block = 32.
__global__ void pack_frags_kernel(const float* __restrict__ w, __bf16* __restrict__ pk,
                                  int K, int N) {
  long fid = blockIdx.x;
  int ntiles = N >> 4;
  long fpm = (long)(K >> 5) * ntiles;
  long m = fid / fpm;
  long f = fid % fpm;
  int kc = (int)(f / ntiles), nt = (int)(f % ntiles);
  int l = threadIdx.x;
  int g = l >> 4, c = l & 15;
  const float* wm = w + m * (long)K * N;
  __bf16* dst = pk + (fid * 32 + l) * 16;
#pragma unroll
  for (int e = 0; e < 16; ++e) {
    int kpat = (e < 8 ? e : e + 8) + 8 * g;
    dst[e] = (__bf16)wm[(long)(kc * 32 + kpat) * N + nt * 16 + c];
  }
}

// ---------------- kernel 1: LayerNorm ----------------
// block = 128 threads, one row of 512 per block; contiguous float4 per thread.
__global__ void ln_kernel(const float* __restrict__ x, const float* __restrict__ gamma,
                          const float* __restrict__ beta, float* __restrict__ xn,
                          __bf16* __restrict__ xn_bf) {
  long row = blockIdx.x;
  int t = threadIdx.x;
  v4f v = *(const v4f*)(x + row * H_ + t * 4);
  float s = v[0] + v[1] + v[2] + v[3];
  float ss = v[0] * v[0] + v[1] * v[1] + v[2] * v[2] + v[3] * v[3];
#pragma unroll
  for (int off = 16; off > 0; off >>= 1) {
    s += __shfl_xor(s, off, 32);
    ss += __shfl_xor(ss, off, 32);
  }
  __shared__ float rs[4], rss[4];
  int wid = t >> 5;
  if ((t & 31) == 0) { rs[wid] = s; rss[wid] = ss; }
  __syncthreads();
  s = rs[0] + rs[1] + rs[2] + rs[3];
  ss = rss[0] + rss[1] + rss[2] + rss[3];
  float mean = s * (1.f / H_);
  float var = ss * (1.f / H_) - mean * mean;
  float inv = rsqrtf(var + LN_EPS_);
  v4f gm = *(const v4f*)(gamma + t * 4);
  v4f bt = *(const v4f*)(beta + t * 4);
  v4f o;
  v8bf ob;
#pragma unroll
  for (int j = 0; j < 4; ++j) {
    o[j] = (v[j] - mean) * inv * gm[j] + bt[j];
    ob[j] = (__bf16)o[j];
  }
  *(v4f*)(xn + row * H_ + t * 4) = o;
  // store 4 bf16 (8 bytes)
  __bf16* xb = xn_bf + row * H_ + t * 4;
  xb[0] = ob[0]; xb[1] = ob[1]; xb[2] = ob[2]; xb[3] = ob[3];
}

// ---------------- kernel 2: Q/K/V projections ----------------
// grid = (S/32, B*NH, 3), block = 64 (2 waves); wave: 16(S) x 64(DK) tile.
// z=0: q (scaled 1/8, [bn][s][d]); z=1: k ([bn][s][d]); z=2: v transposed ([bn][d][s]).
__global__ void proj_kernel(const __bf16* __restrict__ xn_bf,
                            const __bf16* __restrict__ wq_pk, const __bf16* __restrict__ wk_pk,
                            const __bf16* __restrict__ wv_pk,
                            __bf16* __restrict__ q, __bf16* __restrict__ k,
                            __bf16* __restrict__ vT) {
  int which = blockIdx.z;
  int wid = threadIdx.x >> 5;
  int l = threadIdx.x & 31;
  int g = l >> 4, c = l & 15;
  int bn = blockIdx.y;
  int n = bn & 7;
  long s0 = (long)blockIdx.x * 32 + wid * 16;
  long row0 = ((long)(bn >> 3)) * S_ + s0;
  const __bf16* wsel = (which == 0) ? wq_pk : (which == 1) ? wk_pk : wv_pk;
  const __bf16* wpk = wsel + (long)n * 64 * 512;  // 64 frags/head, 512 bf16/frag

  v8f acc[4];
#pragma unroll
  for (int t = 0; t < 4; ++t) acc[t] = zero_v8f();

  for (int kc = 0; kc < H_ / 32; ++kc) {
    v16bf a = load_frag_rm(xn_bf, row0, H_, kc * 32);
#pragma unroll
    for (int t = 0; t < 4; ++t)
      acc[t] = wmma_bf16(a, load_frag_pk(wpk, kc * 4 + t), acc[t]);
  }

  if (which == 2) {
    // vT[bn][d][s]: lane writes 8 contiguous s per tile
    __bf16* vbase = vT + (long)bn * DK_ * S_;
#pragma unroll
    for (int t = 0; t < 4; ++t) {
      __bf16* p = vbase + (long)(t * 16 + c) * S_ + s0 + 8 * g;
#pragma unroll
      for (int i = 0; i < 8; ++i) p[i] = (__bf16)acc[t][i];
    }
  } else {
    __bf16* dst = (which == 0) ? q : k;
    float sc = (which == 0) ? 0.125f : 1.f;  // fold 1/sqrt(DK) into q
    long base = ((long)bn * S_ + s0) * DK_;
#pragma unroll
    for (int t = 0; t < 4; ++t)
#pragma unroll
      for (int i = 0; i < 8; ++i)
        dst[base + (long)(i + 8 * g) * DK_ + t * 16 + c] = (__bf16)(acc[t][i] * sc);
  }
}

// ---------------- kernel 3: flash-style masked attention ----------------
// grid = (S/64, B*NH), block = 128 (4 waves); wave handles one 16-row q tile.
__global__ void attn_kernel(const __bf16* __restrict__ qm, const __bf16* __restrict__ km,
                            const __bf16* __restrict__ vT, const int* __restrict__ mask,
                            __bf16* __restrict__ h) {
  int wid = threadIdx.x >> 5;
  int l = threadIdx.x & 31;
  int g = l >> 4, c = l & 15;
  int bn = blockIdx.y;
  int b = bn >> 3, n = bn & 7;
  long s0 = (long)blockIdx.x * 64 + wid * 16;
  const __bf16* qb = qm + (long)bn * S_ * DK_;
  const __bf16* kb = km + (long)bn * S_ * DK_;
  const __bf16* vb = vT + (long)bn * DK_ * S_;   // [d][s]
  const int* mrow = mask + (long)b * S_;

  __shared__ __bf16 pbuf[4][16 * 32];
  __bf16* pb = pbuf[wid];

  v16bf aq0 = load_frag_rm(qb, s0, DK_, 0);
  v16bf aq1 = load_frag_rm(qb, s0, DK_, 32);

  v8f acc[4];
  float mr[8], lr[8];
#pragma unroll
  for (int t = 0; t < 4; ++t) acc[t] = zero_v8f();
#pragma unroll
  for (int i = 0; i < 8; ++i) { mr[i] = -1e38f; lr[i] = 0.f; }

  for (int kt = 0; kt < S_; kt += 32) {
    v8f s0f = zero_v8f(), s1f = zero_v8f();
    s0f = wmma_bf16(aq0, load_frag_rm(kb, kt, DK_, 0), s0f);
    s0f = wmma_bf16(aq1, load_frag_rm(kb, kt, DK_, 32), s0f);
    s1f = wmma_bf16(aq0, load_frag_rm(kb, kt + 16, DK_, 0), s1f);
    s1f = wmma_bf16(aq1, load_frag_rm(kb, kt + 16, DK_, 32), s1f);

    bool m0 = mrow[kt + c] != 0;
    bool m1 = mrow[kt + 16 + c] != 0;
    float tm[8];
#pragma unroll
    for (int i = 0; i < 8; ++i) {
      s0f[i] = m0 ? s0f[i] : NEG_INF_;
      s1f[i] = m1 ? s1f[i] : NEG_INF_;
      tm[i] = fmaxf(s0f[i], s1f[i]);
    }
#pragma unroll
    for (int off = 8; off > 0; off >>= 1)
#pragma unroll
      for (int i = 0; i < 8; ++i) tm[i] = fmaxf(tm[i], __shfl_xor(tm[i], off, 32));

    float p0[8], p1[8], psum[8];
#pragma unroll
    for (int i = 0; i < 8; ++i) {
      float mn = fmaxf(mr[i], tm[i]);
      float corr = __expf(mr[i] - mn);
      mr[i] = mn;
      lr[i] *= corr;
      acc[0][i] *= corr; acc[1][i] *= corr; acc[2][i] *= corr; acc[3][i] *= corr;
      p0[i] = __expf(s0f[i] - mn);
      p1[i] = __expf(s1f[i] - mn);
      psum[i] = p0[i] + p1[i];
    }
#pragma unroll
    for (int off = 8; off > 0; off >>= 1)
#pragma unroll
      for (int i = 0; i < 8; ++i) psum[i] += __shfl_xor(psum[i], off, 32);
#pragma unroll
    for (int i = 0; i < 8; ++i) lr[i] += psum[i];

    // probs: C layout -> A layout via per-wave LDS tile (16x32 bf16)
#pragma unroll
    for (int i = 0; i < 8; ++i) {
      pb[(i + 8 * g) * 32 + c] = (__bf16)p0[i];
      pb[(i + 8 * g) * 32 + 16 + c] = (__bf16)p1[i];
    }
    asm volatile("s_wait_dscnt 0" ::: "memory");
    v16bf pa = load_frag_rm(pb, 0, 32, 0);
#pragma unroll
    for (int t = 0; t < 4; ++t)
      acc[t] = wmma_bf16(pa, load_frag_rm(vb, t * 16, S_, kt), acc[t]);
  }

  long hbase = ((long)b * S_ + s0) * H_ + n * DK_;
#pragma unroll
  for (int i = 0; i < 8; ++i) {
    float inv = 1.f / lr[i];
#pragma unroll
    for (int t = 0; t < 4; ++t)
      h[hbase + (long)(i + 8 * g) * H_ + t * 16 + c] = (__bf16)(acc[t][i] * inv);
  }
}

// ---------------- kernel 4: output projection + residual + mask ----------------
// grid = B*S/16, block = 256 (8 waves); wave w computes cols [w*64, w*64+64).
__global__ void outproj_kernel(const __bf16* __restrict__ hbf, const __bf16* __restrict__ wo_pk,
                               const float* __restrict__ xn, const int* __restrict__ mask,
                               float* __restrict__ out) {
  int wid = threadIdx.x >> 5;
  int l = threadIdx.x & 31;
  int g = l >> 4, c = l & 15;
  long row0 = (long)blockIdx.x * 16;
  int n0 = wid * 64;

  v8f acc[4];
#pragma unroll
  for (int t = 0; t < 4; ++t) acc[t] = zero_v8f();

  for (int kc = 0; kc < H_ / 32; ++kc) {
    v16bf a = load_frag_rm(hbf, row0, H_, kc * 32);
#pragma unroll
    for (int t = 0; t < 4; ++t)
      acc[t] = wmma_bf16(a, load_frag_pk(wo_pk, kc * 32 + wid * 4 + t), acc[t]);
  }
#pragma unroll
  for (int i = 0; i < 8; ++i) {
    long row = row0 + i + 8 * g;
    float mf = (mask[row] != 0) ? 1.f : 0.f;
#pragma unroll
    for (int t = 0; t < 4; ++t) {
      long cidx = row * H_ + n0 + t * 16 + c;
      out[cidx] = (acc[t][i] + xn[cidx]) * mf;
    }
  }
}

extern "C" void kernel_launch(void* const* d_in, const int* in_sizes, int n_in,
                              void* d_out, int out_size, void* d_ws, size_t ws_size,
                              hipStream_t stream) {
  (void)in_sizes; (void)n_in; (void)out_size; (void)ws_size;
  const float* x     = (const float*)d_in[0];
  const int*   mask  = (const int*)d_in[1];
  const float* Wq    = (const float*)d_in[2];
  const float* Wk    = (const float*)d_in[3];
  const float* Wv    = (const float*)d_in[4];
  const float* Wo    = (const float*)d_in[5];
  const float* gamma = (const float*)d_in[6];
  const float* beta  = (const float*)d_in[7];
  float* out = (float*)d_out;

  char* ws = (char*)d_ws;
  float* xn = (float*)ws;        ws += (size_t)B_ * S_ * H_ * 4;
  __bf16* xn_bf = (__bf16*)ws;   ws += (size_t)B_ * S_ * H_ * 2;
  __bf16* qb = (__bf16*)ws;      ws += (size_t)B_ * NH_ * S_ * DK_ * 2;
  __bf16* kb = (__bf16*)ws;      ws += (size_t)B_ * NH_ * S_ * DK_ * 2;
  __bf16* vT = (__bf16*)ws;      ws += (size_t)B_ * NH_ * S_ * DK_ * 2;
  __bf16* hb = (__bf16*)ws;      ws += (size_t)B_ * S_ * H_ * 2;
  __bf16* wq_pk = (__bf16*)ws;   ws += (size_t)NH_ * H_ * DK_ * 2;
  __bf16* wk_pk = (__bf16*)ws;   ws += (size_t)NH_ * H_ * DK_ * 2;
  __bf16* wv_pk = (__bf16*)ws;   ws += (size_t)NH_ * H_ * DK_ * 2;
  __bf16* wo_pk = (__bf16*)ws;   ws += (size_t)H_ * H_ * 2;

  // weight packing: per-head KxN = 512x64 -> 8*16*4 = 512 frag-blocks; Wo: 16*32 = 512
  pack_frags_kernel<<<NH_ * (H_ / 32) * (DK_ / 16), 32, 0, stream>>>(Wq, wq_pk, H_, DK_);
  pack_frags_kernel<<<NH_ * (H_ / 32) * (DK_ / 16), 32, 0, stream>>>(Wk, wk_pk, H_, DK_);
  pack_frags_kernel<<<NH_ * (H_ / 32) * (DK_ / 16), 32, 0, stream>>>(Wv, wv_pk, H_, DK_);
  pack_frags_kernel<<<(H_ / 32) * (H_ / 16), 32, 0, stream>>>(Wo, wo_pk, H_, H_);

  ln_kernel<<<B_ * S_, 128, 0, stream>>>(x, gamma, beta, xn, xn_bf);

  proj_kernel<<<dim3(S_ / 32, B_ * NH_, 3), 64, 0, stream>>>(xn_bf, wq_pk, wk_pk, wv_pk,
                                                             qb, kb, vT);

  attn_kernel<<<dim3(S_ / 64, B_ * NH_), 128, 0, stream>>>(qb, kb, vT, mask, hb);

  outproj_kernel<<<B_ * S_ / 16, 256, 0, stream>>>(hb, wo_pk, xn, mask, out);
}
